// NTMCell_28819230556621
// MI455X (gfx1250) — matrix operand order repaired and checked by reading, not compiled, not run on hardware
//
#include <hip/hip_runtime.h>
#include <math.h>

#define BATCH 2048
#define NLOC  256
#define MW    128
#define UDIM  512
#define HEADR 134
#define HEADW 390
#define HSTR  1048          // 2*HEADR + 2*HEADW
#define CLIPV 20.0f

typedef __attribute__((ext_vector_type(16))) _Float16 v16h;
typedef __attribute__((ext_vector_type(8)))  _Float16 v8h;
typedef __attribute__((ext_vector_type(4)))  _Float16 v4h;
typedef __attribute__((ext_vector_type(8)))  float    v8f;
typedef __attribute__((ext_vector_type(4)))  float    v4f;

__device__ __forceinline__ float sigf(float x) { return 1.0f / (1.0f + expf(-x)); }
__device__ __forceinline__ float softplusf(float x) { return (x > 20.0f) ? x : log1pf(expf(x)); }

// ---------------------------------------------------------------------------
// WMMA GEMM: C[M,N] = clip(A@B + bias).  f32 in/out, f16 WMMA compute.
// Assumes M % 64 == 0 and K % 32 == 0 (true for every call site).
// ALIGNED_N: N % 64 == 0 -> no column guards, fully vectorized B staging.
// TRANSB:    B element [k][n] read as Bm[n*ldb+k].
// Block: 256 threads = 8 waves (2x4), block tile 64x64, each wave owns two
// 16x16 output tiles (rows waveM*32+{0,16}), K-step 32, double-buffered LDS
// (single barrier per K-step; staging overlaps ds_load+WMMA of current tile).
// LDS rows stride 40 halves (80B): 16B-aligned chunks, bank-conflict-free.
// ---------------------------------------------------------------------------
template <bool ALIGNED_N, bool TRANSB>
__global__ __launch_bounds__(256)
void wmma_gemm(const float* __restrict__ A, int lda,
               const float* __restrict__ Bm, int ldb,
               float* __restrict__ C, int ldc,
               const float* __restrict__ bias, int doClip,
               int M, int N, int K) {
  __shared__ _Float16 sA[2][64][40];
  __shared__ _Float16 sBt[2][64][40];

  const int tid  = threadIdx.x;
  const int wave = tid >> 5;
  const int lane = tid & 31;
  const int waveM = wave >> 2;        // 0..1
  const int waveN = wave & 3;         // 0..3
  const int blockM = blockIdx.y * 64;
  const int blockN = blockIdx.x * 64;

  auto stage = [&](int buf, int k0) {
    // ---- A tile 64x32 as f16: 512 x v4f, 2 per thread, no guards ----
#pragma unroll
    for (int i = 0; i < 2; ++i) {
      int vidx = i * 256 + tid;            // 0..511
      int r = vidx >> 3, c4 = (vidx & 7) * 4;
      v4f v = *(const v4f*)&A[(size_t)(blockM + r) * lda + (k0 + c4)];
      v4h h;
      h[0] = (_Float16)v[0]; h[1] = (_Float16)v[1];
      h[2] = (_Float16)v[2]; h[3] = (_Float16)v[3];
      *(v4h*)&sA[buf][r][c4] = h;
    }
    // ---- B tile 32x64 transposed into sBt[n][k] ----
    if (ALIGNED_N) {
      if (!TRANSB) {
#pragma unroll
        for (int i = 0; i < 2; ++i) {
          int vidx = i * 256 + tid;        // 0..511
          int kl = vidx >> 4, nl4 = (vidx & 15) * 4;
          v4f v = *(const v4f*)&Bm[(size_t)(k0 + kl) * ldb + (blockN + nl4)];
          sBt[buf][nl4 + 0][kl] = (_Float16)v[0];
          sBt[buf][nl4 + 1][kl] = (_Float16)v[1];
          sBt[buf][nl4 + 2][kl] = (_Float16)v[2];
          sBt[buf][nl4 + 3][kl] = (_Float16)v[3];
        }
      } else {
#pragma unroll
        for (int i = 0; i < 2; ++i) {
          int vidx = i * 256 + tid;
          int nl = vidx >> 3, k4 = (vidx & 7) * 4;
          v4f v = *(const v4f*)&Bm[(size_t)(blockN + nl) * ldb + (k0 + k4)];
          v4h h;
          h[0] = (_Float16)v[0]; h[1] = (_Float16)v[1];
          h[2] = (_Float16)v[2]; h[3] = (_Float16)v[3];
          *(v4h*)&sBt[buf][nl][k4] = h;
        }
      }
    } else {
      // branch-free: clamp index, unconditional load, select zero (TRANSB==0)
#pragma unroll
      for (int i = 0; i < 8; ++i) {
        int idx = i * 256 + tid;           // 0..2047
        int kl = idx >> 6, nl = idx & 63;
        int gn = blockN + nl;
        int gnc = (gn < N) ? gn : (N - 1);
        float raw = Bm[(size_t)(k0 + kl) * ldb + gnc];
        sBt[buf][nl][kl] = (_Float16)((gn < N) ? raw : 0.0f);
      }
    }
  };

  v8f acc0 = {}, acc1 = {};

  const int kbaseA = (lane >= 16) ? 16 : 0;   // A: lanes>=16 hold K 16..31
  const int kbaseB = (lane >= 16) ? 16 : 0;   // B: lanes>=16 hold K 16..31
  const int rowA = waveM * 32 + (lane & 15);
  const int colB = waveN * 16 + (lane & 15);

  stage(0, 0);
  int buf = 0;
  for (int k0 = 0; k0 < K; k0 += 32) {
    __syncthreads();                          // staging of `buf` complete
    if (k0 + 32 < K) stage(buf ^ 1, k0 + 32); // prefetch next tile into other buffer

    v8h a0lo = *(const v8h*)&sA[buf][rowA][kbaseA];
    v8h a0hi = *(const v8h*)&sA[buf][rowA][kbaseA + 8];
    v8h a1lo = *(const v8h*)&sA[buf][rowA + 16][kbaseA];
    v8h a1hi = *(const v8h*)&sA[buf][rowA + 16][kbaseA + 8];
    v16h bfrag = *(const v16h*)&sBt[buf][colB][kbaseB];
    v16h af0 = __builtin_shufflevector(a0lo, a0hi, 0,1,2,3,4,5,6,7,8,9,10,11,12,13,14,15);
    v16h af1 = __builtin_shufflevector(a1lo, a1hi, 0,1,2,3,4,5,6,7,8,9,10,11,12,13,14,15);

    acc0 = __builtin_amdgcn_wmma_f32_16x16x32_f16(false, af0, false, bfrag, (short)0, acc0, false, false);
    acc1 = __builtin_amdgcn_wmma_f32_16x16x32_f16(false, af1, false, bfrag, (short)0, acc1, false, false);
    buf ^= 1;
  }

  // ---- store: C layout VGPR r -> row r + 8*(lane>=16); lane&15 -> col ----
  const int gcol = blockN + waveN * 16 + (lane & 15);
  if (ALIGNED_N || gcol < N) {
    float bv = bias ? bias[gcol] : 0.0f;
    const int rbase = blockM + waveM * 32 + ((lane >= 16) ? 8 : 0);
#pragma unroll
    for (int r = 0; r < 8; ++r) {
      float v0 = acc0[r] + bv;
      float v1 = acc1[r] + bv;
      if (doClip) {
        v0 = fminf(fmaxf(v0, -CLIPV), CLIPV);
        v1 = fminf(fmaxf(v1, -CLIPV), CLIPV);
      }
      C[(size_t)(rbase + r) * ldc + gcol]      = v0;
      C[(size_t)(rbase + 16 + r) * ldc + gcol] = v1;
    }
  }
}

// ---------------------------------------------------------------------------
// Batch-constant precompute #1
// ---------------------------------------------------------------------------
__global__ void pre1_kernel(const float* __restrict__ R0, const float* __restrict__ W_prep,
                            const float* __restrict__ b_prep, const float* __restrict__ A0,
                            const float* __restrict__ m0,
                            float* h_pre, float* A_prev, float* m_norm) {
  __shared__ float rc[256];
  int t = threadIdx.x;                 // 0..511
  if (t < 256) rc[t] = tanhf(R0[t]);
  __syncthreads();
  float acc = b_prep[t];
  for (int j = 0; j < 256; ++j) acc += rc[j] * W_prep[j * UDIM + t];
  h_pre[t] = acc;
  if (t < NLOC) {
    float s = 0.0f;
    for (int j = 0; j < MW; ++j) { float v = m0[t * MW + j]; s += v * v; }
    m_norm[t] = sqrtf(s);
  }
  if (t < 4) {
    const float* row = A0 + t * NLOC;
    float mx = row[0];
    for (int j = 1; j < NLOC; ++j) mx = fmaxf(mx, row[j]);
    float s = 0.0f;
    for (int j = 0; j < NLOC; ++j) s += expf(row[j] - mx);
    for (int j = 0; j < NLOC; ++j) A_prev[t * NLOC + j] = expf(row[j] - mx) / s;
  }
}

// ---------------------------------------------------------------------------
// Batch-constant precompute #2: per-column LSTM constants
// ---------------------------------------------------------------------------
__global__ void pre2_kernel(const float* __restrict__ W_lstm, const float* __restrict__ b_lstm,
                            const float* __restrict__ H0, const float* __restrict__ h_pre,
                            float* z_const0, float* z_const1) {
  int c = blockIdx.x * blockDim.x + threadIdx.x;
  if (c >= 4 * UDIM) return;
  const float* W0 = W_lstm;
  const float* W1 = W_lstm + (size_t)3 * UDIM * 4 * UDIM;
  float z0 = b_lstm[c];
  float z1 = b_lstm[4 * UDIM + c];
  for (int u = 0; u < UDIM; ++u) {
    z0 += h_pre[u]     * W0[(size_t)(UDIM + u)     * (4 * UDIM) + c];
    z0 += H0[u]        * W0[(size_t)(2 * UDIM + u) * (4 * UDIM) + c];
    z1 += H0[UDIM + u] * W1[(size_t)(2 * UDIM + u) * (4 * UDIM) + c];
  }
  z_const0[c] = z0;
  z_const1[c] = z1;
}

// ---------------------------------------------------------------------------
// LSTM gate epilogue
// ---------------------------------------------------------------------------
__global__ void gates_kernel(const float* __restrict__ z, const float* __restrict__ zc,
                             const float* __restrict__ c0, float* __restrict__ hout,
                             int doClip) {
  int idx = blockIdx.x * blockDim.x + threadIdx.x;
  if (idx >= BATCH * UDIM) return;
  int b = idx / UDIM, u = idx - b * UDIM;
  const float* zr = z + (size_t)b * 4 * UDIM;
  float ig = zr[u]            + zc[u];
  float fg = zr[UDIM + u]     + zc[UDIM + u];
  float gg = zr[2 * UDIM + u] + zc[2 * UDIM + u];
  float og = zr[3 * UDIM + u] + zc[3 * UDIM + u];
  float c  = sigf(fg) * c0[u] + sigf(ig) * tanhf(gg);
  float h  = sigf(og) * tanhf(c);
  if (doClip) h = fminf(fmaxf(h, -CLIPV), CLIPV);
  hout[idx] = h;
}

// concat [X | h1] -> xh[B,1024]
__global__ void catxh_kernel(const float* __restrict__ X, const float* __restrict__ h1,
                             float* __restrict__ xh) {
  int idx = blockIdx.x * blockDim.x + threadIdx.x;
  if (idx >= BATCH * 1024) return;
  int b = idx >> 10, c = idx & 1023;
  xh[idx] = (c < UDIM) ? X[(size_t)b * UDIM + c] : h1[(size_t)b * UDIM + (c - UDIM)];
}

// k_all[head-major row][128] = tanh(key), knorm = |k|
__global__ void makek_kernel(const float* __restrict__ heads,
                             float* __restrict__ k_all, float* __restrict__ knorm) {
  __shared__ float red[128];
  int idx = blockIdx.x;                  // i*BATCH + b
  int i = idx >> 11, b = idx & (BATCH - 1);
  int base = (i < 2) ? i * HEADR : 2 * HEADR + (i - 2) * HEADW;
  int t = threadIdx.x;                   // 128
  float kv = tanhf(heads[(size_t)b * HSTR + base + t]);
  k_all[(size_t)idx * MW + t] = kv;
  red[t] = kv * kv;
  __syncthreads();
  for (int s = 64; s > 0; s >>= 1) { if (t < s) red[t] += red[t + s]; __syncthreads(); }
  if (t == 0) knorm[idx] = sqrtf(red[0]);
}

// ---------------------------------------------------------------------------
// Addressing: cosine softmax -> interpolate -> circular shift -> sharpen
// ---------------------------------------------------------------------------
__global__ void addr_kernel(const float* __restrict__ heads, const float* __restrict__ inner,
                            const float* __restrict__ knorm, const float* __restrict__ m_norm,
                            const float* __restrict__ A_prev, float* __restrict__ A_t) {
  __shared__ float red[256];
  __shared__ float wg[256];
  int idx = blockIdx.x;
  int i = idx >> 11, b = idx & (BATCH - 1);
  int base = (i < 2) ? i * HEADR : 2 * HEADR + (i - 2) * HEADW;
  const float* hd = heads + (size_t)b * HSTR + base;
  float beta  = softplusf(hd[MW]);
  float g     = sigf(hd[MW + 1]);
  float s0r = hd[MW + 2], s1r = hd[MW + 3], s2r = hd[MW + 4];
  float smx = fmaxf(s0r, fmaxf(s1r, s2r));
  float e0 = expf(s0r - smx), e1 = expf(s1r - smx), e2 = expf(s2r - smx);
  float es = e0 + e1 + e2;
  float s0 = e0 / es, s1 = e1 / es, s2 = e2 / es;
  float gamma = softplusf(hd[MW + 5]);

  int n = threadIdx.x;
  float Kv = inner[(size_t)idx * NLOC + n] / (knorm[idx] * m_norm[n] + 1e-8f);
  float x  = beta * Kv;

  red[n] = x; __syncthreads();
  for (int s = 128; s > 0; s >>= 1) { if (n < s) red[n] = fmaxf(red[n], red[n + s]); __syncthreads(); }
  float mx = red[0]; __syncthreads();
  float ex = expf(x - mx);
  red[n] = ex; __syncthreads();
  for (int s = 128; s > 0; s >>= 1) { if (n < s) red[n] += red[n + s]; __syncthreads(); }
  float ssum = red[0]; __syncthreads();
  float wc = ex / ssum;

  float wgv = g * wc + (1.0f - g) * A_prev[i * NLOC + n];
  wg[n] = wgv; __syncthreads();
  float wconv = s0 * wg[n] + s1 * wg[(n + NLOC - 1) & (NLOC - 1)] + s2 * wg[(n + 1) & (NLOC - 1)];
  float wsh = powf(wconv, gamma);
  red[n] = wsh; __syncthreads();
  for (int s = 128; s > 0; s >>= 1) { if (n < s) red[n] += red[n + s]; __syncthreads(); }
  A_t[(size_t)idx * NLOC + n] = wsh / red[0];
}

// erase/add vectors per (write-head,batch)
__global__ void da_kernel(const float* __restrict__ heads,
                          float* __restrict__ dbuf, float* __restrict__ abuf) {
  int idx = blockIdx.x;                  // i*BATCH + b
  int i = idx >> 11, b = idx & (BATCH - 1);
  int base = 2 * HEADR + i * HEADW;
  int t = threadIdx.x;                   // 128
  dbuf[(size_t)idx * MW + t] = sigf(heads[(size_t)b * HSTR + base + HEADR + t]);
  abuf[(size_t)idx * MW + t] = tanhf(heads[(size_t)b * HSTR + base + HEADR + MW + t]);
}

// yin = [h | R0 | R1]
__global__ void yin_kernel(const float* __restrict__ hbuf, const float* __restrict__ Rbuf,
                           float* __restrict__ yin) {
  int idx = blockIdx.x * blockDim.x + threadIdx.x;
  if (idx >= BATCH * 768) return;
  int b = idx / 768, c = idx - b * 768;
  float v;
  if (c < UDIM)           v = hbuf[(size_t)b * UDIM + c];
  else if (c < UDIM + MW) v = Rbuf[(size_t)b * MW + (c - UDIM)];
  else                    v = Rbuf[(size_t)(BATCH + b) * MW + (c - UDIM - MW)];
  yin[idx] = v;
}

// ---------------------------------------------------------------------------
// Fused double erase/add memory write: 268 MB streaming store, v4f + NT hints.
// 256 threads handle 8 (b,n) rows of 128 per block.
// ---------------------------------------------------------------------------
__global__ __launch_bounds__(256)
void mt_kernel(const float* __restrict__ m0, const float* __restrict__ A_t,
               const float* __restrict__ dbuf, const float* __restrict__ abuf,
               float* __restrict__ out) {
  int row = blockIdx.x * 8 + (threadIdx.x >> 5);   // row = b*NLOC + n
  int b = row >> 8, n = row & (NLOC - 1);
  int j4 = (threadIdx.x & 31) * 4;
  float w0 = A_t[(size_t)(2 * BATCH + b) * NLOC + n];
  float w1 = A_t[(size_t)(3 * BATCH + b) * NLOC + n];
  v4f mv = *(const v4f*)&m0[n * MW + j4];
  v4f d0 = *(const v4f*)&dbuf[(size_t)b * MW + j4];
  v4f a0 = *(const v4f*)&abuf[(size_t)b * MW + j4];
  v4f d1 = *(const v4f*)&dbuf[(size_t)(BATCH + b) * MW + j4];
  v4f a1 = *(const v4f*)&abuf[(size_t)(BATCH + b) * MW + j4];
  v4f v = mv * (1.0f - w0 * d0) + w0 * a0;
  v = v * (1.0f - w1 * d1) + w1 * a1;
  __builtin_nontemporal_store(v, (v4f*)&out[(size_t)BATCH * UDIM + (size_t)row * MW + j4]);
}

// ---------------------------------------------------------------------------
extern "C" void kernel_launch(void* const* d_in, const int* in_sizes, int n_in,
                              void* d_out, int out_size, void* d_ws, size_t ws_size,
                              hipStream_t stream) {
  (void)in_sizes; (void)n_in; (void)out_size; (void)ws_size;
  const float* X      = (const float*)d_in[0];
  const float* m0     = (const float*)d_in[1];
  const float* H0     = (const float*)d_in[2];
  const float* C0     = (const float*)d_in[3];
  const float* R0     = (const float*)d_in[4];
  const float* A0     = (const float*)d_in[5];
  const float* W_prep = (const float*)d_in[6];
  const float* b_prep = (const float*)d_in[7];
  const float* W_lstm = (const float*)d_in[8];
  const float* b_lstm = (const float*)d_in[9];
  const float* W_r    = (const float*)d_in[10];
  const float* b_r    = (const float*)d_in[11];
  const float* W_w    = (const float*)d_in[12];
  const float* b_w    = (const float*)d_in[13];
  const float* W_ou   = (const float*)d_in[14];
  const float* b_ou   = (const float*)d_in[15];
  float* out = (float*)d_out;
  float* ws  = (float*)d_ws;

  size_t o = 0;
  float* h_pre    = ws + o; o += 512;
  float* z_const0 = ws + o; o += 2048;
  float* z_const1 = ws + o; o += 2048;
  float* A_prev   = ws + o; o += 4 * NLOC;
  float* m_norm   = ws + o; o += NLOC;
  float* knorm    = ws + o; o += 4 * BATCH;
  float* zbuf     = ws + o; o += (size_t)BATCH * 4 * UDIM;
  float* h1       = ws + o; o += (size_t)BATCH * UDIM;
  float* hbuf     = ws + o; o += (size_t)BATCH * UDIM;
  float* xh       = ws + o; o += (size_t)BATCH * 1024;
  float* heads    = ws + o; o += (size_t)BATCH * HSTR;
  float* k_all    = ws + o; o += (size_t)4 * BATCH * MW;
  float* inner    = ws + o; o += (size_t)4 * BATCH * NLOC;
  float* A_t      = ws + o; o += (size_t)4 * BATCH * NLOC;
  float* Rbuf     = ws + o; o += (size_t)2 * BATCH * MW;
  float* yin      = ws + o; o += (size_t)BATCH * 768;
  float* dbuf     = ws + o; o += (size_t)2 * BATCH * MW;
  float* abuf     = ws + o; o += (size_t)2 * BATCH * MW;

  auto gemm = [&](const float* A, int lda, const float* Bm, int ldb, int transB,
                  float* C, int ldc, const float* bias, int doClip,
                  int M, int N, int K) {
    dim3 grid((N + 63) / 64, M / 64);
    if ((N & 63) == 0) {
      if (transB)
        wmma_gemm<true, true><<<grid, 256, 0, stream>>>(A, lda, Bm, ldb, C, ldc, bias, doClip, M, N, K);
      else
        wmma_gemm<true, false><<<grid, 256, 0, stream>>>(A, lda, Bm, ldb, C, ldc, bias, doClip, M, N, K);
    } else {
      wmma_gemm<false, false><<<grid, 256, 0, stream>>>(A, lda, Bm, ldb, C, ldc, bias, doClip, M, N, K);
    }
  };

  // batch-constant precompute
  pre1_kernel<<<1, 512, 0, stream>>>(R0, W_prep, b_prep, A0, m0, h_pre, A_prev, m_norm);
  pre2_kernel<<<8, 256, 0, stream>>>(W_lstm, b_lstm, H0, h_pre, z_const0, z_const1);

  // LSTM layer 0
  gemm(X, UDIM, W_lstm, 4 * UDIM, 0, zbuf, 4 * UDIM, nullptr, 0, BATCH, 4 * UDIM, UDIM);
  gates_kernel<<<(BATCH * UDIM) / 256, 256, 0, stream>>>(zbuf, z_const0, C0, h1, 0);

  // LSTM layer 1
  catxh_kernel<<<(BATCH * 1024) / 256, 256, 0, stream>>>(X, h1, xh);
  gemm(xh, 1024, W_lstm + (size_t)3 * UDIM * 4 * UDIM, 4 * UDIM, 0,
       zbuf, 4 * UDIM, nullptr, 0, BATCH, 4 * UDIM, 1024);
  gates_kernel<<<(BATCH * UDIM) / 256, 256, 0, stream>>>(zbuf, z_const1, C0 + UDIM, hbuf, 1);

  // head projections into one [B,1048] buffer
  gemm(hbuf, UDIM, W_r, 2 * HEADR, 0, heads, HSTR, b_r, 0, BATCH, 2 * HEADR, UDIM);
  gemm(hbuf, UDIM, W_w, 2 * HEADW, 0, heads + 2 * HEADR, HSTR, b_w, 0, BATCH, 2 * HEADW, UDIM);

  // content addressing: stacked k @ m0^T
  makek_kernel<<<4 * BATCH, 128, 0, stream>>>(heads, k_all, knorm);
  gemm(k_all, MW, m0, MW, 1, inner, NLOC, nullptr, 0, 4 * BATCH, NLOC, MW);
  addr_kernel<<<4 * BATCH, NLOC, 0, stream>>>(heads, inner, knorm, m_norm, A_prev, A_t);

  // reads: A_t[read heads] @ m0
  gemm(A_t, NLOC, m0, MW, 0, Rbuf, MW, nullptr, 0, 2 * BATCH, MW, NLOC);

  // output projection
  yin_kernel<<<(BATCH * 768) / 256, 256, 0, stream>>>(hbuf, Rbuf, yin);
  gemm(yin, 768, W_ou, UDIM, 0, out, UDIM, b_ou, 1, BATCH, UDIM, 768);

  // memory update (bandwidth-dominant, NT streaming stores)
  da_kernel<<<2 * BATCH, 128, 0, stream>>>(heads, dbuf, abuf);
  mt_kernel<<<(BATCH * NLOC) / 8, 256, 0, stream>>>(m0, A_t, dbuf, abuf, out);
}
